// Attention_18141941858635
// MI455X (gfx1250) — matrix operand order repaired and checked by reading, not compiled, not run on hardware
//
#include <hip/hip_runtime.h>
#include <hip/hip_bf16.h>
#include <math.h>

// Problem constants (B,S,HID)=(2,2048,2048), (NH,NKV,HD)=(16,4,128)
#define B_SZ   2
#define S_LEN  2048
#define HID    2048
#define NH     16
#define NKV    4
#define HDIM   128
#define QKVH   (NH + 2*NKV)      // 24 heads in fused qkv
#define QKV_LD (QKVH * HDIM)     // 3072 elements per (b,s) row

typedef __attribute__((ext_vector_type(16))) _Float16 v16h;
typedef __attribute__((ext_vector_type(8)))  _Float16 v8h;
typedef __attribute__((ext_vector_type(4)))  _Float16 v4h;
typedef __attribute__((ext_vector_type(8)))  float    v8f;
typedef __attribute__((ext_vector_type(4)))  unsigned int u32x4;
typedef __attribute__((ext_vector_type(8)))  int          i32x8;
typedef __attribute__((ext_vector_type(4)))  int          i32x4;

#if defined(__has_builtin)
#  if __has_builtin(__builtin_amdgcn_tensor_load_to_lds)
#    define HAVE_TDM 1
#  else
#    define HAVE_TDM 0
#  endif
#else
#  define HAVE_TDM 0
#endif

// ---------------------------------------------------------------------------
// WMMA wrapper: D = A(16x32 f16) * B(32x16 f16) + C(16x16 f32)
// ---------------------------------------------------------------------------
__device__ __forceinline__ v8f wmma_f16(v16h a, v16h b, v8f c) {
  return __builtin_amdgcn_wmma_f32_16x16x32_f16(
      false, a, false, b, (short)0, c, false, false);
}

// ---------------------------------------------------------------------------
// f16 fragment loaders (work for global or LDS pointers; all 16B aligned).
// A-frag (16x32): lanes 0-15 -> row=lane, K=0..15;
//                 lanes 16-31 -> row=lane-16, K=16..31.
// ---------------------------------------------------------------------------
__device__ __forceinline__ v16h load_a_frag_h(const _Float16* base, int ld) {
  const int lane = threadIdx.x & 31;
  const int row  = lane & 15;
  const int koff = (lane >> 4) * 16;
  v8h lo = *reinterpret_cast<const v8h*>(base + (size_t)row * ld + koff);
  v8h hi = *reinterpret_cast<const v8h*>(base + (size_t)row * ld + koff + 8);
  v16h r;
#pragma unroll
  for (int i = 0; i < 8; ++i) { r[i] = lo[i]; r[8 + i] = hi[i]; }
  return r;
}

// B-frag (32x16) from row-major [n, k] (k contiguous): lanes 0-15 -> col=lane,
// K=0..15; lanes 16-31 -> col=lane-16, K=16..31.
__device__ __forceinline__ v16h load_b_frag_h(const _Float16* base, int ld) {
  const int lane = threadIdx.x & 31;
  const int col  = lane & 15;
  const int koff = (lane >> 4) * 16;
  const v8h* p = reinterpret_cast<const v8h*>(base + (size_t)col * ld + koff);
  v8h lo = p[0], hi = p[1];
  v16h r;
#pragma unroll
  for (int i = 0; i < 8; ++i) { r[i] = lo[i]; r[8 + i] = hi[i]; }
  return r;
}

// B-frag (32x16) with strided K (V matrix: rows = keys, cols = head-dim).
__device__ __forceinline__ v16h load_b_frag_col_h(const _Float16* base, int ld) {
  const int lane  = threadIdx.x & 31;
  const int col   = lane & 15;
  const int kbase = (lane >> 4) * 16;
  v16h r;
#pragma unroll
  for (int i = 0; i < 16; ++i)
    r[i] = base[(size_t)(kbase + i) * ld + col];
  return r;
}

#if HAVE_TDM
__device__ __forceinline__ unsigned rfl(unsigned x) {
  return (unsigned)__builtin_amdgcn_readfirstlane((int)x);
}
// ---------------------------------------------------------------------------
// TENSOR_LOAD_TO_LDS of a 2D tile (tile_w x tile_h elements) from a strided
// row-major region into contiguous LDS. dsize_code: 1 = 2B elems, 2 = 4B.
// All descriptor words forced to SGPRs via readfirstlane (wave-uniform).
// ---------------------------------------------------------------------------
__device__ __forceinline__ void tdm_load_2d(
    const void* gaddr, unsigned lds_byte, unsigned dsize_code,
    unsigned tile_w, unsigned tile_h, unsigned long long row_stride) {
  unsigned long long ga = (unsigned long long)(size_t)gaddr;
  u32x4 g0;
  g0[0] = rfl(1u);                                       // count=1 (valid D#)
  g0[1] = rfl(lds_byte);                                 // lds_addr
  g0[2] = rfl((unsigned)ga);                             // global_addr[31:0]
  g0[3] = rfl((unsigned)((ga >> 32) & 0x01FFFFFFu)       // global_addr[56:32]
              | (2u << 30));                             // type=2 ("image")
  i32x8 g1;
  g1[0] = (int)rfl(dsize_code << 16);                    // data_size, mask=0
  g1[1] = (int)rfl((tile_w & 0xFFFFu) << 16);            // tensor_dim0[15:0]
  g1[2] = (int)rfl(((tile_w >> 16) & 0xFFFFu)            // tensor_dim0[31:16]
              | ((tile_h & 0xFFFFu) << 16));             // tensor_dim1[15:0]
  g1[3] = (int)rfl(((tile_h >> 16) & 0xFFFFu)            // tensor_dim1[31:16]
              | ((tile_w & 0xFFFFu) << 16));             // tile_dim0
  g1[4] = (int)rfl(tile_h & 0xFFFFu);                    // tile_dim1; dim2=0
  g1[5] = (int)rfl((unsigned)(row_stride & 0xFFFFFFFFull));   // d0_stride lo
  g1[6] = (int)rfl((unsigned)((row_stride >> 32) & 0xFFFFull)); // d0_stride hi
  g1[7] = 0;
  i32x4 z4 = {0, 0, 0, 0};
#if __clang_major__ >= 23
  i32x8 z8 = {0, 0, 0, 0, 0, 0, 0, 0};
  __builtin_amdgcn_tensor_load_to_lds(g0, g1, z4, z4, z8, 0);
#else
  __builtin_amdgcn_tensor_load_to_lds(g0, g1, z4, z4, 0);
#endif
}
#endif // HAVE_TDM

// ---------------------------------------------------------------------------
// f32 -> f16 streaming convert, 4 elems/thread (sizes all divide 1024).
// ---------------------------------------------------------------------------
__global__ __launch_bounds__(256) void cvt_f32_f16(const float* __restrict__ in,
                                                   _Float16* __restrict__ outp) {
  size_t i = ((size_t)blockIdx.x * 256 + threadIdx.x) * 4;
  float4 v = *reinterpret_cast<const float4*>(in + i);
  v4h h;
  h[0] = (_Float16)v.x; h[1] = (_Float16)v.y;
  h[2] = (_Float16)v.z; h[3] = (_Float16)v.w;
  *reinterpret_cast<v4h*>(outp + i) = h;
}

// ---------------------------------------------------------------------------
// GEMM (NT): C[M,N] = A[M,K] * W[N,K]^T, f16 inputs, f32 out, f16 WMMA.
// Block = 256 threads = 8 waves (4x2), block tile 128x64, wave tile 32x32.
// TDM path: double-buffered f16 LDS tiles, wave0 drives the Tensor Data
// Mover, pipelined with s_wait_tensorcnt; inner loop is ds_load + wmma only.
// ---------------------------------------------------------------------------
__global__ __launch_bounds__(256) void gemm_nt_wmma(const _Float16* __restrict__ A,
                                                    const _Float16* __restrict__ W,
                                                    float* __restrict__ C,
                                                    int M, int N, int K) {
  const int wave = threadIdx.x >> 5;
  const int lane = threadIdx.x & 31;
  const int wm = wave & 3;
  const int wn = wave >> 2;
  const int m0 = blockIdx.x * 128;
  const int n0 = blockIdx.y * 64;

  v8f acc00 = {}, acc01 = {}, acc10 = {}, acc11 = {};

#if HAVE_TDM
  __shared__ _Float16 sA[2][128][32];   // 2 x 8 KB
  __shared__ _Float16 sW[2][64][32];    // 2 x 4 KB
  const unsigned ldsA0 = (unsigned)(size_t)(void*)&sA[0][0][0];
  const unsigned ldsA1 = (unsigned)(size_t)(void*)&sA[1][0][0];
  const unsigned ldsW0 = (unsigned)(size_t)(void*)&sW[0][0][0];
  const unsigned ldsW1 = (unsigned)(size_t)(void*)&sW[1][0][0];
  const int nk = K >> 5;

  if (threadIdx.x < 32) {               // wave 0 drives the TDM
    tdm_load_2d(A + (size_t)m0 * K, ldsA0, 1u, 32u, 128u, (unsigned long long)K);
    tdm_load_2d(W + (size_t)n0 * K, ldsW0, 1u, 32u, 64u,  (unsigned long long)K);
  }

  for (int i = 0; i < nk; ++i) {
    if (threadIdx.x < 32) {
      if (i + 1 < nk) {
        const int k1 = (i + 1) << 5;
        tdm_load_2d(A + (size_t)m0 * K + k1, ((i + 1) & 1) ? ldsA1 : ldsA0,
                    1u, 32u, 128u, (unsigned long long)K);
        tdm_load_2d(W + (size_t)n0 * K + k1, ((i + 1) & 1) ? ldsW1 : ldsW0,
                    1u, 32u, 64u, (unsigned long long)K);
        __builtin_amdgcn_s_wait_tensorcnt(2);   // stage i landed
      } else {
        __builtin_amdgcn_s_wait_tensorcnt(0);   // final stage landed
      }
    }
    __syncthreads();                            // publish buf[i&1]

    const int buf = i & 1;
    v16h a0 = load_a_frag_h(&sA[buf][wm * 32][0], 32);
    v16h a1 = load_a_frag_h(&sA[buf][wm * 32 + 16][0], 32);
    v16h b0 = load_b_frag_h(&sW[buf][wn * 32][0], 32);
    v16h b1 = load_b_frag_h(&sW[buf][wn * 32 + 16][0], 32);
    acc00 = wmma_f16(a0, b0, acc00);
    acc01 = wmma_f16(a0, b1, acc01);
    acc10 = wmma_f16(a1, b0, acc10);
    acc11 = wmma_f16(a1, b1, acc11);

    __syncthreads();                            // WAR before buf re-fill
  }
#else
  for (int k0 = 0; k0 < K; k0 += 32) {
    v16h a0 = load_a_frag_h(A + (size_t)(m0 + wm * 32) * K + k0, K);
    v16h a1 = load_a_frag_h(A + (size_t)(m0 + wm * 32 + 16) * K + k0, K);
    v16h b0 = load_b_frag_h(W + (size_t)(n0 + wn * 32) * K + k0, K);
    v16h b1 = load_b_frag_h(W + (size_t)(n0 + wn * 32 + 16) * K + k0, K);
    acc00 = wmma_f16(a0, b0, acc00);
    acc01 = wmma_f16(a0, b1, acc01);
    acc10 = wmma_f16(a1, b0, acc10);
    acc11 = wmma_f16(a1, b1, acc11);
  }
#endif

  const int mw = m0 + wm * 32;
  const int nw = n0 + wn * 32;
  const int col   = lane & 15;
  const int rbase = (lane >> 4) * 8;
#pragma unroll
  for (int r = 0; r < 8; ++r) {
    C[(size_t)(mw + rbase + r) * N + (nw + col)]           = acc00[r];
    C[(size_t)(mw + rbase + r) * N + (nw + 16 + col)]      = acc01[r];
    C[(size_t)(mw + 16 + rbase + r) * N + (nw + col)]      = acc10[r];
    C[(size_t)(mw + 16 + rbase + r) * N + (nw + 16 + col)] = acc11[r];
  }
}

// ---------------------------------------------------------------------------
// RoPE on q (heads 0..15) and k (heads 16..19) in the fused f32 qkv buffer.
// ---------------------------------------------------------------------------
__global__ __launch_bounds__(256) void rope_kernel(float* __restrict__ qkv,
                                                   const float* __restrict__ cosT,
                                                   const float* __restrict__ sinT) {
  int idx = blockIdx.x * blockDim.x + threadIdx.x;
  int d   = idx & 63;
  int t   = idx >> 6;
  int hh  = t % (NH + NKV);
  int s   = (t / (NH + NKV)) % S_LEN;
  int b   = t / ((NH + NKV) * S_LEN);

  float c0 = cosT[(size_t)s * HDIM + d];
  float s0 = sinT[(size_t)s * HDIM + d];
  float c1 = cosT[(size_t)s * HDIM + d + 64];
  float s1 = sinT[(size_t)s * HDIM + d + 64];

  float* p = qkv + (((size_t)b * S_LEN + s) * QKVH + hh) * HDIM;
  float x1 = p[d];
  float x2 = p[d + 64];
  p[d]      = x1 * c0 - x2 * s0;
  p[d + 64] = x2 * c1 + x1 * s1;
}

// ---------------------------------------------------------------------------
// Causal GQA flash attention, f16 qkv in / f16 attn out.
// Grid: (S/128, NH, B). Block: 256 = 8 waves; each wave owns 16 query rows
// and streams 32-key tiles to the diagonal. Per-wave TDM stages K and V
// tiles into private LDS (TENSORcnt is per-wave -> no cross-wave sync);
// K waited at tensorcnt<=1 so V transfer overlaps QK^T + softmax.
// ---------------------------------------------------------------------------
__global__ __launch_bounds__(256) void flash_attn_wmma(const _Float16* __restrict__ qkv,
                                                       _Float16* __restrict__ aout) {
  __shared__ _Float16 Pbuf[8][16 * 32];          // 8 KB
#if HAVE_TDM
  __shared__ _Float16 Kbuf[8][32 * 128];         // 64 KB (8 KB / wave)
  __shared__ _Float16 Vbuf[8][32 * 128];         // 64 KB
#endif

  const int wave  = threadIdx.x >> 5;
  const int lane  = threadIdx.x & 31;
  const int h     = blockIdx.y;
  const int b     = blockIdx.z;
  const int g     = h >> 2;                      // kv head (n_rep = 4)
  const int qbase = blockIdx.x * 128 + wave * 16;

  const _Float16* qp = qkv + ((size_t)b * S_LEN * QKVH + h) * HDIM;
  const _Float16* kp = qkv + ((size_t)b * S_LEN * QKVH + NH + g) * HDIM;
  const _Float16* vp = qkv + ((size_t)b * S_LEN * QKVH + NH + NKV + g) * HDIM;

  v16h qf[4];
#pragma unroll
  for (int ks = 0; ks < 4; ++ks)
    qf[ks] = load_a_frag_h(qp + (size_t)qbase * QKV_LD + ks * 32, QKV_LD);

  v8f o[8] = {};
  float mrow[8], lrow[8];
#pragma unroll
  for (int r = 0; r < 8; ++r) { mrow[r] = -1e30f; lrow[r] = 0.0f; }

  const float sscale = 0.08838834764831845f;     // 1/sqrt(128), applied to scores
  const int colk  = lane & 15;
  const int rbase = (lane >> 4) * 8;
  _Float16* P = &Pbuf[wave][0];
#if HAVE_TDM
  const unsigned kb_lds = (unsigned)(size_t)(void*)&Kbuf[wave][0];
  const unsigned vb_lds = (unsigned)(size_t)(void*)&Vbuf[wave][0];
  const _Float16* kb = &Kbuf[wave][0];
  const _Float16* vb = &Vbuf[wave][0];
#endif

  for (int j0 = 0; j0 <= qbase + 15; j0 += 32) {
#if HAVE_TDM
    // WAR: all LDS reads of the previous tile retired before TDM re-fills it.
    asm volatile("s_wait_dscnt 0x0" ::: "memory");
    tdm_load_2d(kp + (size_t)j0 * QKV_LD, kb_lds, 1u, 128u, 32u,
                (unsigned long long)QKV_LD);
    tdm_load_2d(vp + (size_t)j0 * QKV_LD, vb_lds, 1u, 128u, 32u,
                (unsigned long long)QKV_LD);
    __builtin_amdgcn_s_wait_tensorcnt(1);        // K landed; V still in flight
    __builtin_amdgcn_fence(__ATOMIC_ACQUIRE, "workgroup");
#endif

    // ---- S = Q K^T (two 16-key N tiles, K=128 over 4 steps) ----
    v8f s0 = {}, s1 = {};
#pragma unroll
    for (int ks = 0; ks < 4; ++ks) {
#if HAVE_TDM
      v16h kb0 = load_b_frag_h(kb + ks * 32, 128);
      v16h kb1 = load_b_frag_h(kb + 16 * 128 + ks * 32, 128);
#else
      v16h kb0 = load_b_frag_h(kp + (size_t)j0 * QKV_LD + ks * 32, QKV_LD);
      v16h kb1 = load_b_frag_h(kp + (size_t)(j0 + 16) * QKV_LD + ks * 32, QKV_LD);
#endif
      s0 = wmma_f16(qf[ks], kb0, s0);
      s1 = wmma_f16(qf[ks], kb1, s1);
    }

    // ---- scale + causal mask + per-row tile max ----
    float pm[8];
#pragma unroll
    for (int r = 0; r < 8; ++r) {
      const int row = qbase + rbase + r;
      float x0 = s0[r] * sscale, x1 = s1[r] * sscale;
      if (j0 + colk > row)      x0 = -1e30f;
      if (j0 + 16 + colk > row) x1 = -1e30f;
      s0[r] = x0; s1[r] = x1;
      pm[r] = fmaxf(x0, x1);
    }
#pragma unroll
    for (int mk = 1; mk < 16; mk <<= 1)
#pragma unroll
      for (int r = 0; r < 8; ++r)
        pm[r] = fmaxf(pm[r], __shfl_xor(pm[r], mk, 32));

    // ---- online softmax update ----
    float corr[8], psum[8];
#pragma unroll
    for (int r = 0; r < 8; ++r) {
      float mn = fmaxf(mrow[r], pm[r]);
      corr[r]  = __expf(mrow[r] - mn);
      mrow[r]  = mn;
      float p0 = __expf(s0[r] - mn);
      float p1 = __expf(s1[r] - mn);
      s0[r] = p0; s1[r] = p1;
      psum[r] = p0 + p1;
    }
#pragma unroll
    for (int mk = 1; mk < 16; mk <<= 1)
#pragma unroll
      for (int r = 0; r < 8; ++r)
        psum[r] += __shfl_xor(psum[r], mk, 32);
#pragma unroll
    for (int r = 0; r < 8; ++r)
      lrow[r] = lrow[r] * corr[r] + psum[r];
#pragma unroll
    for (int t = 0; t < 8; ++t)
#pragma unroll
      for (int r = 0; r < 8; ++r)
        o[t][r] *= corr[r];

    // ---- stage P (16x32 f16) in per-wave LDS to rebuild as A-fragment ----
#pragma unroll
    for (int r = 0; r < 8; ++r) {
      P[(rbase + r) * 32 + colk]      = (_Float16)s0[r];
      P[(rbase + r) * 32 + 16 + colk] = (_Float16)s1[r];
    }
    __builtin_amdgcn_fence(__ATOMIC_RELEASE, "wavefront");
    __builtin_amdgcn_wave_barrier();
    __builtin_amdgcn_fence(__ATOMIC_ACQUIRE, "wavefront");

    v16h pf;
    {
      const int prow = lane & 15;
      const int pk   = (lane >> 4) * 16;
#pragma unroll
      for (int i = 0; i < 8; ++i) pf[i]     = P[prow * 32 + pk + i];
#pragma unroll
      for (int i = 0; i < 8; ++i) pf[8 + i] = P[prow * 32 + pk + 8 + i];
    }

#if HAVE_TDM
    __builtin_amdgcn_s_wait_tensorcnt(0);        // V landed
    __builtin_amdgcn_fence(__ATOMIC_ACQUIRE, "workgroup");
#endif

    // ---- O += P V (8 head-dim N tiles) ----
#pragma unroll
    for (int t = 0; t < 8; ++t) {
#if HAVE_TDM
      v16h vbf = load_b_frag_col_h(vb + t * 16, 128);
#else
      v16h vbf = load_b_frag_col_h(vp + (size_t)j0 * QKV_LD + t * 16, QKV_LD);
#endif
      o[t] = wmma_f16(pf, vbf, o[t]);
    }
    __builtin_amdgcn_wave_barrier();             // WAR on P before next store
  }

  // ---- normalize and store f16 [b, s, h*128 + n] ----
#pragma unroll
  for (int r = 0; r < 8; ++r) {
    float inv = 1.0f / lrow[r];
    const size_t row = (size_t)b * S_LEN + qbase + rbase + r;
#pragma unroll
    for (int t = 0; t < 8; ++t)
      aout[row * (NH * HDIM) + h * HDIM + t * 16 + colk] =
          (_Float16)(o[t][r] * inv);
  }
}

// ---------------------------------------------------------------------------
// Launch: cvt(hidden,Wqkv,Wo) -> f16 GEMM -> RoPE -> cvt(qkv) -> flash ->
// f16 GEMM. Workspace ~130 MB.
// ---------------------------------------------------------------------------
extern "C" void kernel_launch(void* const* d_in, const int* in_sizes, int n_in,
                              void* d_out, int out_size, void* d_ws, size_t ws_size,
                              hipStream_t stream) {
  const float* hidden = (const float*)d_in[0];
  const float* cosT   = (const float*)d_in[1];
  const float* sinT   = (const float*)d_in[2];
  const float* Wqkv   = (const float*)d_in[3];
  const float* Wo     = (const float*)d_in[4];
  float* out = (float*)d_out;

  const size_t M   = (size_t)B_SZ * S_LEN;       // 4096
  const size_t nQ  = M * QKV_LD;                 // 12.58M (qkv elems)
  const size_t nH  = M * HID;                    // 8.39M (hidden / attn elems)
  const size_t nWq = (size_t)QKV_LD * HID;       // 6.29M
  const size_t nWo = (size_t)HID * HID;          // 4.19M

  char* ws = (char*)d_ws;
  float*     qkv   = (float*)ws;                 ws += nQ  * sizeof(float);
  _Float16*  hh    = (_Float16*)ws;              ws += nH  * sizeof(_Float16);
  _Float16*  wqh   = (_Float16*)ws;              ws += nWq * sizeof(_Float16);
  _Float16*  woh   = (_Float16*)ws;              ws += nWo * sizeof(_Float16);
  _Float16*  qkvh  = (_Float16*)ws;              ws += nQ  * sizeof(_Float16);
  _Float16*  attnh = (_Float16*)ws;

  dim3 blk(256);

  // one-time f32 -> f16 conversions (streamed, 4 elems/thread)
  cvt_f32_f16<<<dim3((unsigned)(nH  / 1024)), blk, 0, stream>>>(hidden, hh);
  cvt_f32_f16<<<dim3((unsigned)(nWq / 1024)), blk, 0, stream>>>(Wqkv, wqh);
  cvt_f32_f16<<<dim3((unsigned)(nWo / 1024)), blk, 0, stream>>>(Wo, woh);

  // qkv = hidden @ W_qkv^T : M=4096, N=3072, K=2048 (f32 accum out)
  gemm_nt_wmma<<<dim3(M / 128, QKV_LD / 64), blk, 0, stream>>>(
      hh, wqh, qkv, (int)M, QKV_LD, HID);

  // RoPE on q and k heads (f32)
  const int rope_threads = B_SZ * S_LEN * (NH + NKV) * (HDIM / 2);
  rope_kernel<<<dim3(rope_threads / 256), blk, 0, stream>>>(qkv, cosT, sinT);

  // qkv -> f16 for attention
  cvt_f32_f16<<<dim3((unsigned)(nQ / 1024)), blk, 0, stream>>>(qkv, qkvh);

  // causal GQA attention (f16 in/out)
  flash_attn_wmma<<<dim3(S_LEN / 128, NH, B_SZ), blk, 0, stream>>>(qkvh, attnh);

  // out = attn @ W_o^T : M=4096, N=2048, K=2048 (f32 out)
  gemm_nt_wmma<<<dim3(M / 128, HID / 64), blk, 0, stream>>>(
      attnh, woh, out, (int)M, HID, NH * HDIM);
}
